// MoEFFN_56891136803347
// MI455X (gfx1250) — matrix operand order repaired and checked by reading, not compiled, not run on hardware
//
#include <hip/hip_runtime.h>
#include <stdint.h>

#define D_MODEL 1024
#define D_FFN   4096
#define TWO_FFN 8192
#define NEXP    8
#define TOPK    2
#define NTOK    8192
#define NSLOT   (NTOK * TOPK)   // 16384
#define CAP     1280
#define EPS_R   1e-9f
#define AUXW    0.01f

typedef __attribute__((ext_vector_type(16))) __bf16 v16bf;
typedef __attribute__((ext_vector_type(8)))  float  v8f;
typedef int v4i __attribute__((vector_size(16)));

union Frag { uint4 q[2]; v16bf v; };

__device__ __forceinline__ uint16_t f32_to_bf16(float f) {
  uint32_t u = __builtin_bit_cast(uint32_t, f);
  uint32_t r = (u + 0x7FFFu + ((u >> 16) & 1u)) >> 16;
  return (uint16_t)r;
}

// ---- CDNA5 async global->LDS staging (guarded; falls back to vector copy) ----
#if __has_builtin(__builtin_amdgcn_global_load_async_to_lds_b128)
#define USE_ASYNC 1
#else
#define USE_ASYNC 0
#endif

__device__ __forceinline__ void stage16(const uint32_t* gsrc, uint32_t* ldst) {
#if USE_ASYNC
  __builtin_amdgcn_global_load_async_to_lds_b128(
      (__attribute__((address_space(1))) v4i*)(uintptr_t)gsrc,
      (__attribute__((address_space(3))) v4i*)(uintptr_t)ldst, 0, 0);
#else
  *(uint4*)ldst = *(const uint4*)gsrc;
#endif
}

__device__ __forceinline__ void async_wait() {
#if USE_ASYNC
#if __has_builtin(__builtin_amdgcn_s_wait_asynccnt)
  __builtin_amdgcn_s_wait_asynccnt(0);
#else
  asm volatile("s_wait_asynccnt 0x0" ::: "memory");
#endif
#endif
}

// ---------------- utility kernels ----------------

__global__ void moe_zero_f32(float* p, long n) {
  long i = (long)blockIdx.x * 256 + threadIdx.x;
  long stride = (long)gridDim.x * 256;
  for (; i < n; i += stride) p[i] = 0.f;
}

__global__ void moe_init_route(int* etok, float* egate, float* mP, int* fcnt) {
  int i = blockIdx.x * 256 + threadIdx.x;
  if (i < NEXP) { mP[i] = 0.f; fcnt[i] = 0; }
  if (i < NEXP * CAP) { etok[i] = -1; egate[i] = 0.f; }
}

// fp32 -> packed bf16 pairs (round-nearest-even)
__global__ void moe_cvt_bf16(const float* __restrict__ src, uint32_t* __restrict__ dst, long npairs) {
  long i = (long)blockIdx.x * 256 + threadIdx.x;
  long stride = (long)gridDim.x * 256;
  for (; i < npairs; i += stride) {
    float a = src[2 * i], b = src[2 * i + 1];
    dst[i] = (uint32_t)f32_to_bf16(a) | ((uint32_t)f32_to_bf16(b) << 16);
  }
}

// ---------------- router: one wave per token ----------------

__global__ void moe_router(const float* __restrict__ x, const float* __restrict__ Wr,
                           const float* __restrict__ br, float* __restrict__ gates,
                           int* __restrict__ assign, float* mP, int* fcnt) {
  int lane = threadIdx.x & 31;
  int tok  = blockIdx.x * 8 + (threadIdx.x >> 5);
  float acc[NEXP];
#pragma unroll
  for (int e = 0; e < NEXP; e++) acc[e] = 0.f;
  const float* xr = x + (long)tok * D_MODEL;
  for (int i = 0; i < D_MODEL / 32; i++) {
    int d = i * 32 + lane;
    float xv = xr[d];
#pragma unroll
    for (int e = 0; e < NEXP; e++) acc[e] += xv * Wr[d * NEXP + e];
  }
#pragma unroll
  for (int e = 0; e < NEXP; e++)
#pragma unroll
    for (int off = 16; off >= 1; off >>= 1) acc[e] += __shfl_xor(acc[e], off, 32);

  if (lane == 0) {
    float P[NEXP];
    float mx = -1e30f;
#pragma unroll
    for (int e = 0; e < NEXP; e++) { P[e] = acc[e] + br[e]; mx = fmaxf(mx, P[e]); }
    float s = 0.f;
#pragma unroll
    for (int e = 0; e < NEXP; e++) { P[e] = __expf(P[e] - mx); s += P[e]; }
    float inv = 1.f / s;
#pragma unroll
    for (int e = 0; e < NEXP; e++) { P[e] *= inv; atomicAdd(&mP[e], P[e]); }
    int e0 = 0;
#pragma unroll
    for (int e = 1; e < NEXP; e++) if (P[e] > P[e0]) e0 = e;
    int e1 = (e0 == 0) ? 1 : 0;
#pragma unroll
    for (int e = 0; e < NEXP; e++) if (e != e0 && P[e] > P[e1]) e1 = e;
    float v0 = P[e0], v1 = P[e1];
    float den = v0 + v1 + EPS_R;
    gates[tok * 2]     = v0 / den; assign[tok * 2]     = e0;
    gates[tok * 2 + 1] = v1 / den; assign[tok * 2 + 1] = e1;
    atomicAdd(&fcnt[e0], 1);
    atomicAdd(&fcnt[e1], 1);
  }
}

// ---------------- exact per-expert rank (matches top_k tie-break) ----------------

__global__ void moe_rank(const float* __restrict__ gates, const int* __restrict__ assign,
                         int* __restrict__ rank) {
  __shared__ float gs[256];
  __shared__ int   as[256];
  int i = blockIdx.x * 256 + threadIdx.x;
  float gi = gates[i];
  int   ei = assign[i];
  int   r = 0;
  for (int t = 0; t < NSLOT / 256; t++) {
    int j0 = t * 256;
    gs[threadIdx.x] = gates[j0 + threadIdx.x];
    as[threadIdx.x] = assign[j0 + threadIdx.x];
    __syncthreads();
    for (int j = 0; j < 256; j++) {
      int jj = j0 + j;
      if (as[j] == ei && (gs[j] > gi || (gs[j] == gi && jj < i))) r++;
    }
    __syncthreads();
  }
  rank[i] = r;
}

__global__ void moe_slot(const float* __restrict__ gates, const int* __restrict__ assign,
                         const int* __restrict__ rank, int* __restrict__ etok,
                         float* __restrict__ egate) {
  int i = blockIdx.x * 256 + threadIdx.x;
  if (i >= NSLOT) return;
  int r = rank[i];
  if (r < CAP) {
    int e = assign[i];
    etok[e * CAP + r]  = i / TOPK;
    egate[e * CAP + r] = gates[i];
  }
}

// gather routed tokens into dense bf16 per-expert buffers
__global__ void moe_gather(const float* __restrict__ x, const int* __restrict__ etok,
                           uint16_t* __restrict__ xe) {
  int row = blockIdx.x;  // e*CAP + c
  int tok = etok[row];
  uint16_t* dst = xe + (long)row * D_MODEL;
  if (tok >= 0) {
    const float* src = x + (long)tok * D_MODEL;
    for (int d = threadIdx.x; d < D_MODEL; d += 256) dst[d] = f32_to_bf16(src[d]);
  } else {
    for (int d = threadIdx.x; d < D_MODEL; d += 256) dst[d] = 0;
  }
}

// ---------------- GEMM1 + SwiGLU ----------------
// block tile 64(M) x 128(N of F) for both v and g halves; 8 waves = 2(M) x 4(N),
// wave tile 32x32 -> 8 WMMAs per 32-deep K step from 2 A-frags + 4 B-frags.

__global__ void __launch_bounds__(256) moe_gemm1(const uint32_t* __restrict__ xe32,
                                                 const uint32_t* __restrict__ win32,
                                                 const float* __restrict__ b_in,
                                                 uint16_t* __restrict__ uout) {
  __shared__ __align__(16) uint32_t As[64 * 16];    // 64 rows x 32 K (bf16 pairs)
  __shared__ __align__(16) uint16_t Bv[128 * 32];   // [col][k]
  __shared__ __align__(16) uint16_t Bg[128 * 32];
  int e  = blockIdx.z;
  int n0 = blockIdx.x * 128;  // in [0, D_FFN)
  int m0 = blockIdx.y * 64;
  int tid = threadIdx.x, wave = tid >> 5, lane = tid & 31;
  int wm = wave >> 2, wn = wave & 3;
  const uint32_t* A = xe32 + (long)e * CAP * (D_MODEL / 2);
  const uint32_t* W = win32 + (long)e * D_MODEL * (TWO_FFN / 2);
  v8f accv[2][2] = {};
  v8f accg[2][2] = {};
  int arow_s = tid >> 2;          // A staging: row
  int ac4    = (tid & 3) * 4;     // A staging: u32 quad offset

  for (int kk = 0; kk < D_MODEL; kk += 32) {
    stage16(&A[(long)(m0 + arow_s) * (D_MODEL / 2) + (kk >> 1) + ac4],
            &As[arow_s * 16 + ac4]);
#pragma unroll
    for (int i = 0; i < 8; i++) {
      int f = tid + 256 * i;
      int kr = f >> 6, c2 = f & 63;
      long rowoff = (long)(kk + kr) * (TWO_FFN / 2);
      uint32_t uv = W[rowoff + (n0 >> 1) + c2];
      uint32_t ug = W[rowoff + ((n0 + D_FFN) >> 1) + c2];
      if (i == 0 && kk + 32 < D_MODEL)
        __builtin_prefetch(&W[rowoff + 16 * (TWO_FFN / 2) + (n0 >> 1) + c2], 0, 1);
      Bv[(2 * c2) * 32 + kr]     = (uint16_t)uv;
      Bv[(2 * c2 + 1) * 32 + kr] = (uint16_t)(uv >> 16);
      Bg[(2 * c2) * 32 + kr]     = (uint16_t)ug;
      Bg[(2 * c2 + 1) * 32 + kr] = (uint16_t)(ug >> 16);
    }
    async_wait();
    __syncthreads();

    Frag a[2], bvf[2], bgf[2];
#pragma unroll
    for (int mi = 0; mi < 2; mi++) {
      int ar = wm * 32 + mi * 16 + (lane & 15);
      const uint32_t* ap = &As[ar * 16 + ((lane < 16) ? 0 : 4)];
      a[mi].q[0] = *(const uint4*)ap;       // K pairs 0..3  (this half)
      a[mi].q[1] = *(const uint4*)(ap + 8); // K pairs 8..11 (this half)
    }
    const uint32_t* Bv32 = (const uint32_t*)Bv;
    const uint32_t* Bg32 = (const uint32_t*)Bg;
#pragma unroll
    for (int ni = 0; ni < 2; ni++) {
      int col  = wn * 32 + ni * 16 + (lane & 15);
      int koff = (lane < 16) ? 0 : 4;
      const uint32_t* bp = &Bv32[col * 16 + koff];
      bvf[ni].q[0] = *(const uint4*)bp;
      bvf[ni].q[1] = *(const uint4*)(bp + 8);
      const uint32_t* gp = &Bg32[col * 16 + koff];
      bgf[ni].q[0] = *(const uint4*)gp;
      bgf[ni].q[1] = *(const uint4*)(gp + 8);
    }
#pragma unroll
    for (int mi = 0; mi < 2; mi++)
#pragma unroll
      for (int ni = 0; ni < 2; ni++) {
        accv[mi][ni] = __builtin_amdgcn_wmma_f32_16x16x32_bf16(
            false, a[mi].v, false, bvf[ni].v, (short)0, accv[mi][ni], false, false);
        accg[mi][ni] = __builtin_amdgcn_wmma_f32_16x16x32_bf16(
            false, a[mi].v, false, bgf[ni].v, (short)0, accg[mi][ni], false, false);
      }
    __syncthreads();
  }

#pragma unroll
  for (int mi = 0; mi < 2; mi++) {
    int mbase = m0 + wm * 32 + mi * 16 + ((lane < 16) ? 0 : 8);
#pragma unroll
    for (int ni = 0; ni < 2; ni++) {
      int ncol = n0 + wn * 32 + ni * 16 + (lane & 15);
      float bvb = b_in[e * TWO_FFN + ncol];
      float bgb = b_in[e * TWO_FFN + D_FFN + ncol];
#pragma unroll
      for (int r = 0; r < 8; r++) {
        float v = accv[mi][ni][r] + bvb;
        float g = accg[mi][ni][r] + bgb;
        float u = v * g * (1.f / (1.f + __expf(-g)));
        uout[(long)(e * CAP + mbase + r) * D_FFN + ncol] = f32_to_bf16(u);
      }
    }
  }
}

// ---------------- GEMM2 + gate-scale + scatter-add ----------------
// block tile 64(M) x 128(N of D_MODEL); wave tile 32x32 -> 4 WMMAs per K step.

__global__ void __launch_bounds__(256) moe_gemm2(const uint32_t* __restrict__ u32in,
                                                 const uint32_t* __restrict__ wout32,
                                                 const float* __restrict__ b_out,
                                                 const int* __restrict__ etok,
                                                 const float* __restrict__ egate,
                                                 float* __restrict__ out) {
  __shared__ __align__(16) uint32_t As[64 * 16];
  __shared__ __align__(16) uint16_t Bs[128 * 32];
  int e  = blockIdx.z;
  int n0 = blockIdx.x * 128;  // in [0, D_MODEL)
  int m0 = blockIdx.y * 64;
  int tid = threadIdx.x, wave = tid >> 5, lane = tid & 31;
  int wm = wave >> 2, wn = wave & 3;
  const uint32_t* A = u32in + (long)e * CAP * (D_FFN / 2);
  const uint32_t* W = wout32 + (long)e * D_FFN * (D_MODEL / 2);
  v8f acc[2][2] = {};
  int arow_s = tid >> 2;
  int ac4    = (tid & 3) * 4;

  for (int kk = 0; kk < D_FFN; kk += 32) {
    stage16(&A[(long)(m0 + arow_s) * (D_FFN / 2) + (kk >> 1) + ac4],
            &As[arow_s * 16 + ac4]);
#pragma unroll
    for (int i = 0; i < 8; i++) {
      int f = tid + 256 * i;
      int kr = f >> 6, c2 = f & 63;
      long rowoff = (long)(kk + kr) * (D_MODEL / 2);
      uint32_t uw = W[rowoff + (n0 >> 1) + c2];
      if (i == 0 && kk + 32 < D_FFN)
        __builtin_prefetch(&W[rowoff + 16 * (D_MODEL / 2) + (n0 >> 1) + c2], 0, 1);
      Bs[(2 * c2) * 32 + kr]     = (uint16_t)uw;
      Bs[(2 * c2 + 1) * 32 + kr] = (uint16_t)(uw >> 16);
    }
    async_wait();
    __syncthreads();

    Frag a[2], b[2];
#pragma unroll
    for (int mi = 0; mi < 2; mi++) {
      int ar = wm * 32 + mi * 16 + (lane & 15);
      const uint32_t* ap = &As[ar * 16 + ((lane < 16) ? 0 : 4)];
      a[mi].q[0] = *(const uint4*)ap;
      a[mi].q[1] = *(const uint4*)(ap + 8);
    }
    const uint32_t* Bs32 = (const uint32_t*)Bs;
#pragma unroll
    for (int ni = 0; ni < 2; ni++) {
      int col  = wn * 32 + ni * 16 + (lane & 15);
      int koff = (lane < 16) ? 0 : 4;
      const uint32_t* bp = &Bs32[col * 16 + koff];
      b[ni].q[0] = *(const uint4*)bp;
      b[ni].q[1] = *(const uint4*)(bp + 8);
    }
#pragma unroll
    for (int mi = 0; mi < 2; mi++)
#pragma unroll
      for (int ni = 0; ni < 2; ni++)
        acc[mi][ni] = __builtin_amdgcn_wmma_f32_16x16x32_bf16(
            false, a[mi].v, false, b[ni].v, (short)0, acc[mi][ni], false, false);
    __syncthreads();
  }

#pragma unroll
  for (int mi = 0; mi < 2; mi++) {
    int mbase = m0 + wm * 32 + mi * 16 + ((lane < 16) ? 0 : 8);
#pragma unroll
    for (int ni = 0; ni < 2; ni++) {
      int ncol = n0 + wn * 32 + ni * 16 + (lane & 15);
      float bb = b_out[e * D_MODEL + ncol];
#pragma unroll
      for (int r = 0; r < 8; r++) {
        int slot = e * CAP + mbase + r;
        int tok  = etok[slot];
        if (tok >= 0) {
          float y = (acc[mi][ni][r] + bb) * egate[slot];
          atomicAdd(&out[(long)tok * D_MODEL + ncol], y);
        }
      }
    }
  }
}

__global__ void moe_aux(const float* mP, const int* fcnt, float* out_aux) {
  if (threadIdx.x == 0 && blockIdx.x == 0) {
    float s = 0.f;
    for (int e = 0; e < NEXP; e++) {
      float m = mP[e] / (float)NTOK;
      float f = (float)fcnt[e] / (float)NTOK;
      s += f * m;
    }
    out_aux[0] = (float)NEXP * s * AUXW;
  }
}

// ---------------- host launch ----------------

extern "C" void kernel_launch(void* const* d_in, const int* in_sizes, int n_in,
                              void* d_out, int out_size, void* d_ws, size_t ws_size,
                              hipStream_t stream) {
  const float* x     = (const float*)d_in[0];
  const float* Wr    = (const float*)d_in[1];
  const float* br    = (const float*)d_in[2];
  const float* W_in  = (const float*)d_in[3];
  const float* b_in  = (const float*)d_in[4];
  const float* W_out = (const float*)d_in[5];
  const float* b_out = (const float*)d_in[6];
  float* out = (float*)d_out;

  char* ws = (char*)d_ws;
  size_t off = 0;
  auto take = [&](size_t bytes) { char* p = ws + off; off = (off + bytes + 255) & ~(size_t)255; return p; };

  uint32_t* winB   = (uint32_t*)take((size_t)NEXP * D_MODEL * TWO_FFN * 2);  // bf16
  uint32_t* woutB  = (uint32_t*)take((size_t)NEXP * D_FFN * D_MODEL * 2);    // bf16
  uint16_t* xe     = (uint16_t*)take((size_t)NEXP * CAP * D_MODEL * 2);      // bf16
  uint16_t* ubuf   = (uint16_t*)take((size_t)NEXP * CAP * D_FFN * 2);        // bf16
  float*    gates  = (float*)take((size_t)NSLOT * 4);
  int*      assign = (int*)take((size_t)NSLOT * 4);
  int*      rank   = (int*)take((size_t)NSLOT * 4);
  int*      etok   = (int*)take((size_t)NEXP * CAP * 4);
  float*    egate  = (float*)take((size_t)NEXP * CAP * 4);
  float*    mP     = (float*)take(64);
  int*      fcnt   = (int*)take(64);

  moe_zero_f32<<<2048, 256, 0, stream>>>(out, (long)out_size);
  moe_init_route<<<(NEXP * CAP + 255) / 256, 256, 0, stream>>>(etok, egate, mP, fcnt);
  moe_cvt_bf16<<<4096, 256, 0, stream>>>(W_in, winB, (long)NEXP * D_MODEL * TWO_FFN / 2);
  moe_cvt_bf16<<<4096, 256, 0, stream>>>(W_out, woutB, (long)NEXP * D_FFN * D_MODEL / 2);
  moe_router<<<NTOK / 8, 256, 0, stream>>>(x, Wr, br, gates, assign, mP, fcnt);
  moe_rank<<<NSLOT / 256, 256, 0, stream>>>(gates, assign, rank);
  moe_slot<<<NSLOT / 256, 256, 0, stream>>>(gates, assign, rank, etok, egate);
  moe_gather<<<NEXP * CAP, 256, 0, stream>>>(x, etok, xe);
  moe_gemm1<<<dim3(D_FFN / 128, CAP / 64, NEXP), 256, 0, stream>>>(
      (const uint32_t*)xe, winB, b_in, ubuf);
  moe_gemm2<<<dim3(D_MODEL / 128, CAP / 64, NEXP), 256, 0, stream>>>(
      (const uint32_t*)ubuf, woutB, b_out, etok, egate, out);
  moe_aux<<<1, 64, 0, stream>>>(mP, fcnt, out + (size_t)NTOK * D_MODEL);
}
